// Decoder_71683004171173
// MI455X (gfx1250) — matrix operand order repaired and checked by reading, not compiled, not run on hardware
//
#include <hip/hip_runtime.h>
#include <math.h>

// ---------------------------------------------------------------------------
// Types for CDNA5 WMMA (wave32): D(16x16 f32) = A(16x32 bf16) x B(32x16 bf16) + C
// ---------------------------------------------------------------------------
typedef __attribute__((ext_vector_type(16))) __bf16 v16bf;
typedef __attribute__((ext_vector_type(8)))  float  v8f;
typedef __bf16 bf16;

#define DEVI __device__ __forceinline__

DEVI bf16  f2b(float f) { return (bf16)f; }
DEVI float sigf(float x) { return 1.0f / (1.0f + __expf(-x)); }

// A fragment: row-major [M][ld] bf16 source, tile at (row0, k0).
// Layout (ISA 7.12.2, 16-bit A 16x32): lanes 0-15 -> rows, K = 0..7 & 16..23;
// lanes 16-31 -> rows, K = 8..15 & 24..31.
DEVI v16bf load_a_frag(const bf16* __restrict__ base, int ld, int row0, int k0, int lane) {
  const int half = lane >> 4, r = lane & 15;
  const bf16* p = base + (size_t)(row0 + r) * ld + k0 + half * 8;
  v16bf a;
#pragma unroll
  for (int i = 0; i < 8; ++i) { a[i] = p[i]; a[8 + i] = p[16 + i]; }
  return a;
}

// B fragment from weight matrix W stored row-major [N][K] (i.e. B = W^T).
// Lanes 0-15: col = lane, K=0..7 & 16..23 ; lanes 16-31: col = lane-16, K=8..15 & 24..31.
DEVI v16bf load_b_frag(const bf16* __restrict__ W, int ld, int n0, int k0, int lane) {
  const int half = lane >> 4, c = lane & 15;
  const bf16* p = W + (size_t)(n0 + c) * ld + k0 + half * 8;
  v16bf b;
#pragma unroll
  for (int i = 0; i < 8; ++i) { b[i] = p[i]; b[8 + i] = p[16 + i]; }
  return b;
}

DEVI v8f wmma_bf16(v16bf a, v16bf b, v8f c) {
  // 8 args: (neg_a, A, neg_b, B, c_mod, C, reuse_a, reuse_b)
  return __builtin_amdgcn_wmma_f32_16x16x32_bf16(false, a, false, b, (short)0, c, false, false);
}

// ---------------------------------------------------------------------------
// Problem constants
// ---------------------------------------------------------------------------
#define BATCH 32
#define TENC  512
#define EDIM  512
#define TDEC  800
#define NMEL  80
#define PRE   256
#define ARNN  1024
#define ADIM  128
#define DRNN  1024
#define NFILT 32
#define KSZ   31
#define EMO   256
#define NBLK  32          // persistent blocks (conservative for residency)

// ---------------------------------------------------------------------------
// Pre-processing kernels
// ---------------------------------------------------------------------------
__global__ void pack2_bf16(bf16* __restrict__ dst, const float* __restrict__ s0, int k0,
                           const float* __restrict__ s1, int k1, long total) {
  const long K = k0 + k1;
  const long stride = (long)gridDim.x * blockDim.x;
  for (long i = blockIdx.x * (long)blockDim.x + threadIdx.x; i < total; i += stride) {
    long n = i / K, k = i % K;
    float v = (k < k0) ? s0[n * k0 + k] : s1[n * k1 + (k - k0)];
    dst[i] = f2b(v);
  }
}

__global__ void cvt_pad_bf16(bf16* __restrict__ dst, const float* __restrict__ src,
                             int N, int Kin, int Kout) {
  const long total = (long)N * Kout;
  const long stride = (long)gridDim.x * blockDim.x;
  for (long i = blockIdx.x * (long)blockDim.x + threadIdx.x; i < total; i += stride) {
    int k = (int)(i % Kout);
    long n = i / Kout;
    dst[i] = f2b((k < Kin) ? src[n * Kin + k] : 0.0f);
  }
}

// steps_in: row = t*32 + b ; col k (padded 80->96). t==0 is the go frame (-4.5).
__global__ void build_x0(bf16* __restrict__ X0, const float* __restrict__ din) {
  const long total = (long)TDEC * BATCH * 96;
  const long stride = (long)gridDim.x * blockDim.x;
  for (long i = blockIdx.x * (long)blockDim.x + threadIdx.x; i < total; i += stride) {
    int k = (int)(i % 96);
    long row = i / 96;
    int b = (int)(row % BATCH);
    int t = (int)(row / BATCH);
    float v = 0.0f;
    if (k < NMEL) v = (t == 0) ? -4.5f : din[(size_t)b * NMEL * TDEC + (size_t)k * TDEC + (t - 1)];
    X0[i] = f2b(v);
  }
}

__global__ void init_state(float* __restrict__ zbase, long zwords,
                           float* __restrict__ aw, int* __restrict__ bar) {
  const long stride = (long)gridDim.x * blockDim.x;
  for (long i = blockIdx.x * (long)blockDim.x + threadIdx.x; i < zwords; i += stride)
    zbase[i] = 0.0f;
  for (long i = blockIdx.x * (long)blockDim.x + threadIdx.x; i < BATCH * TENC; i += stride)
    aw[i] = ((i & (TENC - 1)) < 5) ? 0.2f : 0.0f;   // uniform over first 5
  if (blockIdx.x == 0 && threadIdx.x == 0) { bar[0] = 0; bar[1] = 0; }
}

// ---------------------------------------------------------------------------
// Generic WMMA GEMM (one 16x16 tile per wave): out = act( A[MxK] * W[NxK]^T + bias )
// ---------------------------------------------------------------------------
template <bool RELU, bool OUTBF16>
__global__ void gemm16(const bf16* __restrict__ A, const bf16* __restrict__ W,
                       const float* __restrict__ bias, void* __restrict__ out,
                       int M, int N, int K) {
  const int lane = threadIdx.x & 31;
  const int wave = (int)((blockIdx.x * blockDim.x + threadIdx.x) >> 5);
  const int ntiles = N >> 4;
  const int tiles = (M >> 4) * ntiles;
  const int nwaves = (int)((gridDim.x * blockDim.x) >> 5);
  for (int tI = wave; tI < tiles; tI += nwaves) {
    const int row0 = (tI / ntiles) << 4;
    const int n0 = (tI % ntiles) << 4;
    v8f acc = {};
    for (int k0 = 0; k0 < K; k0 += 32) {
      __builtin_prefetch(W + (size_t)n0 * K + k0 + 128, 0, 1);
      v16bf a = load_a_frag(A, K, row0, k0, lane);
      v16bf b = load_b_frag(W, K, n0, k0, lane);
      acc = wmma_bf16(a, b, acc);
    }
    const int half = lane >> 4, c = lane & 15;
    const float bn = bias ? bias[n0 + c] : 0.0f;
#pragma unroll
    for (int i = 0; i < 8; ++i) {
      float v = acc[i] + bn;
      if (RELU) v = fmaxf(v, 0.0f);
      size_t idx = (size_t)(row0 + half * 8 + i) * N + n0 + c;
      if (OUTBF16) ((bf16*)out)[idx] = f2b(v);
      else         ((float*)out)[idx] = v;
    }
  }
}

// ---------------------------------------------------------------------------
// Persistent decoder kernel
// ---------------------------------------------------------------------------
struct Ctx {
  const bf16 *Wa, *Wd;                       // [4096][2048], [4096][2560] bf16
  const float *abih, *abhh, *dbih, *dbhh;    // LSTM biases
  const float *qw;                           // [128][1024]
  const float *convw;                        // [32][2][31]
  const float *densew;                       // [128][32]
  const float *vw, *vb;
  const float *projw, *projb, *gatew, *gateb;
  const float *memory;                       // [32][512][512] f32
  const unsigned char *mask;                 // [32][512] bool
  const bf16 *X;                             // [800*32][256] prenet output
  const bf16 *emoB;                          // [32][256]
  const float *pm;                           // [32*512][128] processed memory
  bf16 *ahB, *ctxB, *hA0, *hA1, *hD0, *hD1;  // bf16 activations (hA/hD ping-pong)
  float *ahf, *dhf, *ctxf, *cA, *cD, *aw, *cum, *loc;
  float *omel, *ogate, *oalign;
  int *bar;
};

__device__ void grid_barrier(int* cnt, int* sense, int* ls) {
  __syncthreads();
  if (threadIdx.x == 0) {
    int s = 1 - *ls;
    *ls = s;
    __threadfence();
    if (__hip_atomic_fetch_add(cnt, 1, __ATOMIC_ACQ_REL, __HIP_MEMORY_SCOPE_AGENT) == NBLK - 1) {
      __hip_atomic_store(cnt, 0, __ATOMIC_RELAXED, __HIP_MEMORY_SCOPE_AGENT);
      __hip_atomic_store(sense, s, __ATOMIC_RELEASE, __HIP_MEMORY_SCOPE_AGENT);
    } else {
      while (__hip_atomic_load(sense, __ATOMIC_ACQUIRE, __HIP_MEMORY_SCOPE_AGENT) != s)
        __builtin_amdgcn_s_sleep(1);
    }
    __threadfence();
  }
  __syncthreads();
}

// Location conv (KSZ=31, pad 15) fused with loc_dense; writes c.loc[32*512][128].
__device__ void conv_loc(const Ctx& c, float* smem) {
  float* cw = smem;           // 32*2*31 = 1984
  float* dw = smem + 1984;    // 128*32 = 4096
  for (int i = threadIdx.x; i < NFILT * 2 * KSZ; i += 256) cw[i] = c.convw[i];
  for (int i = threadIdx.x; i < ADIM * NFILT; i += 256) dw[i] = c.densew[i];
  __syncthreads();
  for (int row = blockIdx.x * 256 + threadIdx.x; row < BATCH * TENC; row += NBLK * 256) {
    const int b = row >> 9, tp = row & (TENC - 1);
    float f[NFILT];
#pragma unroll
    for (int i = 0; i < NFILT; ++i) f[i] = 0.0f;
    const float* aw = c.aw + b * TENC;
    const float* cm = c.cum + b * TENC;
    for (int k = 0; k < KSZ; ++k) {
      const int p = tp + k - (KSZ - 1) / 2;
      const bool ok = (p >= 0) && (p < TENC);
      const float a0 = ok ? aw[p] : 0.0f;
      const float a1 = ok ? cm[p] : 0.0f;
#pragma unroll
      for (int fi = 0; fi < NFILT; ++fi)
        f[fi] += cw[fi * 62 + k] * a0 + cw[fi * 62 + KSZ + k] * a1;
    }
    float* out = c.loc + (size_t)row * ADIM;
    for (int a = 0; a < ADIM; ++a) {
      float v = 0.0f;
#pragma unroll
      for (int fi = 0; fi < NFILT; ++fi) v += dw[a * NFILT + fi] * f[fi];
      out[a] = v;
    }
  }
  __syncthreads();
}

__launch_bounds__(256)
__global__ void decoder_persistent(Ctx c) {
  __shared__ float smem[8192];        // 32 KB: GEMM reduce / attention / conv weights
  __shared__ int localSense;
  if (threadIdx.x == 0) localSense = 0;
  __syncthreads();

  const int lane = threadIdx.x & 31;
  const int wv = threadIdx.x >> 5;

  // Prologue: location features for step 0 (from aw0 / cum=0)
  conv_loc(c, smem);
  grid_barrier(&c.bar[0], &c.bar[1], &localSense);

  for (int t = 0; t < TDEC; ++t) {
    const bf16* hAr = (t & 1) ? c.hA1 : c.hA0;
    bf16*       hAw = (t & 1) ? c.hA0 : c.hA1;
    const bf16* hDr = (t & 1) ? c.hD1 : c.hD0;
    bf16*       hDw = (t & 1) ? c.hD0 : c.hD1;

    // ======== Phase A: attention LSTM (WMMA, K=2048 split into 2 chunks) ========
    {
      const int grp = blockIdx.x * 4 + (wv >> 1);   // 128 tile-groups total
      const int kc = wv & 1;                        // K chunk (1024 each)
      const int row0 = (grp & 1) << 4;              // batch tile
      const int j0 = (grp >> 1) << 4;               // hidden-col tile
      const bf16* Xt = c.X + (size_t)t * BATCH * PRE;
      v8f acc[4] = {};
      // Fixed-bound segment loops: no per-iteration pointer selection -> no
      // exec-mask cascades inside the WMMA K loop.
      auto seg = [&](const bf16* __restrict__ ap, int ld, int aoff, int k0s, int kcnt) {
        for (int kk = 0; kk < kcnt; kk += 32) {
          const int k0 = k0s + kk;
          v16bf a = load_a_frag(ap, ld, row0, aoff + kk, lane);
          __builtin_prefetch(c.Wa + (size_t)j0 * 2048 + k0 + 256, 0, 1);
#pragma unroll
          for (int q = 0; q < 4; ++q) {             // i,f,g,o gates, same hidden cols
            v16bf b = load_b_frag(c.Wa, 2048, q * ARNN + j0, k0, lane);
            acc[q] = wmma_bf16(a, b, acc[q]);
          }
        }
      };
      if (kc == 0) {
        seg(Xt,     256, 0,   0, 256);   // prenet x
        seg(c.ctxB, 512, 0, 256, 512);   // ctx_{t-1}
        seg(c.emoB, 256, 0, 768, 256);   // emotion
      } else {
        seg(hAr,   1024, 0, 1024, 1024); // ah_{t-1}
      }
      const int half = lane >> 4, coln = lane & 15;
#pragma unroll
      for (int q = 0; q < 4; ++q)
#pragma unroll
        for (int i = 0; i < 8; ++i)
          smem[wv * 1024 + q * 256 + (half * 8 + i) * 16 + coln] = acc[q][i];
      __syncthreads();
      // cell update: 4 groups * 256 cells per block
      for (int cell = threadIdx.x; cell < 1024; cell += 256) {
        const int g = cell >> 8, idx = cell & 255;
        const int m = idx >> 4, n = idx & 15;
        const int gg = blockIdx.x * 4 + g;
        const int b = ((gg & 1) << 4) + m;
        const int j = ((gg >> 1) << 4) + n;
        float s[4];
#pragma unroll
        for (int q = 0; q < 4; ++q)
          s[q] = c.abih[q * ARNN + j] + c.abhh[q * ARNN + j]
               + smem[(g * 2 + 0) * 1024 + q * 256 + idx]
               + smem[(g * 2 + 1) * 1024 + q * 256 + idx];
        const float cc = sigf(s[1]) * c.cA[b * ARNN + j] + sigf(s[0]) * tanhf(s[2]);
        const float hh = sigf(s[3]) * tanhf(cc);
        c.cA[b * ARNN + j] = cc;
        c.ahf[b * ARNN + j] = hh;
        const bf16 hb = f2b(hh);
        c.ahB[b * ARNN + j] = hb;   // feeds decoder LSTM A-matrix
        hAw[b * ARNN + j] = hb;     // feeds next step's hh matmul
      }
      __syncthreads();
    }
    grid_barrier(&c.bar[0], &c.bar[1], &localSense);

    // ======== Phase B: attention energies / softmax / context (1 block per b) ====
    if (blockIdx.x < BATCH) {
      const int b = blockIdx.x;
      float* pq  = smem;            // 128
      float* en  = smem + 128;      // 512
      float* red = smem + 768;      // 256
      float* scr = smem + 1024;     // 256
      {
        // pq = ah @ q_w^T : split each K=1024 dot 2-way across 256 threads
        const int a = threadIdx.x >> 1, kh = threadIdx.x & 1;
        const float* ah = c.ahf + b * ARNN + kh * 512;
        const float* qr = c.qw + (size_t)a * ARNN + kh * 512;
        float v = 0.0f;
        for (int k = 0; k < 512; ++k) v += ah[k] * qr[k];
        scr[threadIdx.x] = v;
      }
      __syncthreads();
      if (threadIdx.x < ADIM)
        pq[threadIdx.x] = scr[threadIdx.x * 2] + scr[threadIdx.x * 2 + 1];
      __syncthreads();
      float lmax = -1e30f;
      for (int tt = threadIdx.x; tt < TENC; tt += 256) {
        const float* lc = c.loc + ((size_t)b * TENC + tt) * ADIM;
        const float* pr = c.pm + ((size_t)b * TENC + tt) * ADIM;
        float e = c.vb[0];
        for (int a = 0; a < ADIM; ++a) e += c.vw[a] * tanhf(pq[a] + lc[a] + pr[a]);
        if (c.mask[b * TENC + tt]) e = -10.0f;
        en[tt] = e;
        lmax = fmaxf(lmax, e);
      }
      red[threadIdx.x] = lmax; __syncthreads();
      for (int s = 128; s > 0; s >>= 1) {
        if (threadIdx.x < s) red[threadIdx.x] = fmaxf(red[threadIdx.x], red[threadIdx.x + s]);
        __syncthreads();
      }
      const float mx = red[0]; __syncthreads();
      float lsum = 0.0f;
      for (int tt = threadIdx.x; tt < TENC; tt += 256) {
        const float e = __expf(en[tt] - mx); en[tt] = e; lsum += e;
      }
      red[threadIdx.x] = lsum; __syncthreads();
      for (int s = 128; s > 0; s >>= 1) {
        if (threadIdx.x < s) red[threadIdx.x] += red[threadIdx.x + s];
        __syncthreads();
      }
      const float inv = (1.0f / red[0]) / (1.0f + 1e-8f);  // softmax + renorm
      __syncthreads();
      for (int tt = threadIdx.x; tt < TENC; tt += 256) {
        const float w = en[tt] * inv;
        en[tt] = w;
        c.aw[b * TENC + tt] = w;
        c.cum[b * TENC + tt] += w;
        c.oalign[((size_t)b * TDEC + t) * TENC + tt] = w;
      }
      __syncthreads();
      for (int o = threadIdx.x; o < EDIM; o += 256) {
        const float* mem = c.memory + (size_t)b * TENC * EDIM + o;
        float v = 0.0f;
        for (int tt = 0; tt < TENC; ++tt) v += en[tt] * mem[(size_t)tt * EDIM];
        c.ctxf[b * EDIM + o] = v;
        c.ctxB[b * EDIM + o] = f2b(v);
      }
      __syncthreads();
    }
    grid_barrier(&c.bar[0], &c.bar[1], &localSense);

    // ======== Phase C: decoder LSTM (WMMA, K=2560 split into 2 chunks) ==========
    {
      const int grp = blockIdx.x * 4 + (wv >> 1);
      const int kc = wv & 1;                        // K chunk (1280 each)
      const int row0 = (grp & 1) << 4;
      const int j0 = (grp >> 1) << 4;
      v8f acc[4] = {};
      auto seg = [&](const bf16* __restrict__ ap, int ld, int aoff, int k0s, int kcnt) {
        for (int kk = 0; kk < kcnt; kk += 32) {
          const int k0 = k0s + kk;
          v16bf a = load_a_frag(ap, ld, row0, aoff + kk, lane);
          __builtin_prefetch(c.Wd + (size_t)j0 * 2560 + k0 + 256, 0, 1);
#pragma unroll
          for (int q = 0; q < 4; ++q) {
            v16bf b = load_b_frag(c.Wd, 2560, q * DRNN + j0, k0, lane);
            acc[q] = wmma_bf16(a, b, acc[q]);
          }
        }
      };
      if (kc == 0) {
        seg(c.ahB,  1024,   0,    0, 1024); // ah_t
        seg(c.ctxB,  512,   0, 1024,  256); // ctx_t cols 0..255
      } else {
        seg(c.ctxB,  512, 256, 1280,  256); // ctx_t cols 256..511
        seg(hDr,    1024,   0, 1536, 1024); // dh_{t-1}
      }
      const int half = lane >> 4, coln = lane & 15;
#pragma unroll
      for (int q = 0; q < 4; ++q)
#pragma unroll
        for (int i = 0; i < 8; ++i)
          smem[wv * 1024 + q * 256 + (half * 8 + i) * 16 + coln] = acc[q][i];
      __syncthreads();
      for (int cell = threadIdx.x; cell < 1024; cell += 256) {
        const int g = cell >> 8, idx = cell & 255;
        const int m = idx >> 4, n = idx & 15;
        const int gg = blockIdx.x * 4 + g;
        const int b = ((gg & 1) << 4) + m;
        const int j = ((gg >> 1) << 4) + n;
        float s[4];
#pragma unroll
        for (int q = 0; q < 4; ++q)
          s[q] = c.dbih[q * DRNN + j] + c.dbhh[q * DRNN + j]
               + smem[(g * 2 + 0) * 1024 + q * 256 + idx]
               + smem[(g * 2 + 1) * 1024 + q * 256 + idx];
        const float cc = sigf(s[1]) * c.cD[b * DRNN + j] + sigf(s[0]) * tanhf(s[2]);
        const float hh = sigf(s[3]) * tanhf(cc);
        c.cD[b * DRNN + j] = cc;
        c.dhf[b * DRNN + j] = hh;
        hDw[b * DRNN + j] = f2b(hh);
      }
      __syncthreads();
    }
    grid_barrier(&c.bar[0], &c.bar[1], &localSense);

    // ======== Phase D: projections + location features for next step ============
    if (blockIdx.x < BATCH) {
      const int b = blockIdx.x;
      // dhc = [dh(1024), ctx(512)]; 81 outputs, each K=1536 split 3-way (512 each)
      const int j = threadIdx.x / 3, part = threadIdx.x % 3;
      if (j < NMEL + 1) {
        const float* wrow = (j < NMEL) ? (c.projw + (size_t)j * (DRNN + EDIM)) : c.gatew;
        const float* xsrc; const float* wseg;
        if (part < 2) { xsrc = c.dhf + b * DRNN + part * 512; wseg = wrow + part * 512; }
        else          { xsrc = c.ctxf + b * EDIM;             wseg = wrow + DRNN; }
        float v = 0.0f;
        for (int k = 0; k < 512; ++k) v += xsrc[k] * wseg[k];
        smem[threadIdx.x] = v;
      }
      __syncthreads();
      if (threadIdx.x < NMEL + 1) {
        const int jj = threadIdx.x;
        float v = smem[jj * 3] + smem[jj * 3 + 1] + smem[jj * 3 + 2]
                + ((jj < NMEL) ? c.projb[jj] : c.gateb[0]);
        if (jj < NMEL) c.omel[((size_t)b * NMEL + jj) * TDEC + t] = v;
        else           c.ogate[(size_t)b * TDEC + t] = v;
      }
    }
    __syncthreads();
    if (t + 1 < TDEC) conv_loc(c, smem);   // loc for step t+1 (deps: aw_t, cum_t)
    grid_barrier(&c.bar[0], &c.bar[1], &localSense);
  }
}

// ---------------------------------------------------------------------------
// Host launcher
// ---------------------------------------------------------------------------
extern "C" void kernel_launch(void* const* d_in, const int* in_sizes, int n_in,
                              void* d_out, int out_size, void* d_ws, size_t ws_size,
                              hipStream_t stream) {
  (void)in_sizes; (void)n_in; (void)out_size; (void)ws_size;
  const float* din      = (const float*)d_in[0];
  const float* memory   = (const float*)d_in[1];
  const float* emotion  = (const float*)d_in[2];
  const unsigned char* mask = (const unsigned char*)d_in[3];
  const float* pre_w1   = (const float*)d_in[4];
  const float* pre_b1   = (const float*)d_in[5];
  const float* pre_w2   = (const float*)d_in[6];
  const float* pre_b2   = (const float*)d_in[7];
  const float* arnn_wih = (const float*)d_in[8];
  const float* arnn_whh = (const float*)d_in[9];
  const float* arnn_bih = (const float*)d_in[10];
  const float* arnn_bhh = (const float*)d_in[11];
  const float* q_w      = (const float*)d_in[12];
  const float* conv_w   = (const float*)d_in[13];
  const float* dense_w  = (const float*)d_in[14];
  const float* v_w      = (const float*)d_in[15];
  const float* v_b      = (const float*)d_in[16];
  const float* mem_w    = (const float*)d_in[17];
  const float* mem_b    = (const float*)d_in[18];
  const float* drnn_wih = (const float*)d_in[19];
  const float* drnn_whh = (const float*)d_in[20];
  const float* drnn_bih = (const float*)d_in[21];
  const float* drnn_bhh = (const float*)d_in[22];
  const float* proj_w   = (const float*)d_in[23];
  const float* proj_b   = (const float*)d_in[24];
  const float* gate_w   = (const float*)d_in[25];
  const float* gate_b   = (const float*)d_in[26];

  char* p = (char*)d_ws;
  auto alloc = [&](size_t bytes) -> void* {
    void* r = (void*)p;
    p += (bytes + 255) & ~(size_t)255;
    return r;
  };

  bf16* WaB   = (bf16*)alloc((size_t)4096 * 2048 * 2);
  bf16* WdB   = (bf16*)alloc((size_t)4096 * 2560 * 2);
  bf16* memB  = (bf16*)alloc((size_t)BATCH * TENC * EDIM * 2);
  bf16* memWB = (bf16*)alloc((size_t)ADIM * EDIM * 2);
  bf16* pre1B = (bf16*)alloc((size_t)PRE * 96 * 2);
  bf16* pre2B = (bf16*)alloc((size_t)PRE * PRE * 2);
  bf16* emoB  = (bf16*)alloc((size_t)BATCH * EMO * 2);
  bf16* X0    = (bf16*)alloc((size_t)TDEC * BATCH * 96 * 2);
  bf16* X1    = (bf16*)alloc((size_t)TDEC * BATCH * PRE * 2);
  bf16* X     = (bf16*)alloc((size_t)TDEC * BATCH * PRE * 2);
  float* pm   = (float*)alloc((size_t)BATCH * TENC * ADIM * 4);
  float* loc  = (float*)alloc((size_t)BATCH * TENC * ADIM * 4);
  // zero-initialized state block (all sizes multiples of 256 -> contiguous)
  char* zbeg  = p;
  bf16* ahB   = (bf16*)alloc((size_t)BATCH * ARNN * 2);
  bf16* hA0   = (bf16*)alloc((size_t)BATCH * ARNN * 2);
  bf16* hA1   = (bf16*)alloc((size_t)BATCH * ARNN * 2);
  bf16* hD0   = (bf16*)alloc((size_t)BATCH * DRNN * 2);
  bf16* hD1   = (bf16*)alloc((size_t)BATCH * DRNN * 2);
  bf16* ctxB  = (bf16*)alloc((size_t)BATCH * EDIM * 2);
  float* ahf  = (float*)alloc((size_t)BATCH * ARNN * 4);
  float* dhf  = (float*)alloc((size_t)BATCH * DRNN * 4);
  float* ctxf = (float*)alloc((size_t)BATCH * EDIM * 4);
  float* cA   = (float*)alloc((size_t)BATCH * ARNN * 4);
  float* cD   = (float*)alloc((size_t)BATCH * DRNN * 4);
  float* cum  = (float*)alloc((size_t)BATCH * TENC * 4);
  char* zend  = p;
  float* aw   = (float*)alloc((size_t)BATCH * TENC * 4);
  int* bar    = (int*)alloc(256);

  // --- weight packing / conversion ---
  pack2_bf16<<<2048, 256, 0, stream>>>(WaB, arnn_wih, 1024, arnn_whh, 1024, (long)4096 * 2048);
  pack2_bf16<<<2048, 256, 0, stream>>>(WdB, drnn_wih, 1536, drnn_whh, 1024, (long)4096 * 2560);
  cvt_pad_bf16<<<4096, 256, 0, stream>>>(memB, memory, BATCH * TENC, EDIM, EDIM);
  cvt_pad_bf16<<<64, 256, 0, stream>>>(memWB, mem_w, ADIM, EDIM, EDIM);
  cvt_pad_bf16<<<32, 256, 0, stream>>>(pre1B, pre_w1, PRE, NMEL, 96);
  cvt_pad_bf16<<<64, 256, 0, stream>>>(pre2B, pre_w2, PRE, PRE, PRE);
  cvt_pad_bf16<<<32, 256, 0, stream>>>(emoB, emotion, BATCH, EMO, EMO);
  build_x0<<<1024, 256, 0, stream>>>(X0, din);
  init_state<<<512, 256, 0, stream>>>((float*)zbeg, (long)(zend - zbeg) / 4, aw, bar);

  // --- batched precompute GEMMs (WMMA) ---
  // prenet L1: [25600 x 96] x [256 x 96]^T -> relu -> X1 (bf16)
  gemm16<true, true><<<3200, 256, 0, stream>>>(X0, pre1B, pre_b1, X1, TDEC * BATCH, PRE, 96);
  // prenet L2: [25600 x 256] x [256 x 256]^T -> relu -> X (bf16)
  gemm16<true, true><<<3200, 256, 0, stream>>>(X1, pre2B, pre_b2, X, TDEC * BATCH, PRE, PRE);
  // processed_memory: [16384 x 512] x [128 x 512]^T + mem_b -> pm (f32)
  gemm16<false, false><<<1024, 256, 0, stream>>>(memB, memWB, mem_b, pm, BATCH * TENC, ADIM, EDIM);

  // --- persistent recurrence ---
  float* omel = (float*)d_out;
  float* ogate = omel + (size_t)BATCH * NMEL * TDEC;
  float* oalign = ogate + (size_t)BATCH * TDEC;

  Ctx c;
  c.Wa = WaB; c.Wd = WdB;
  c.abih = arnn_bih; c.abhh = arnn_bhh; c.dbih = drnn_bih; c.dbhh = drnn_bhh;
  c.qw = q_w; c.convw = conv_w; c.densew = dense_w;
  c.vw = v_w; c.vb = v_b;
  c.projw = proj_w; c.projb = proj_b; c.gatew = gate_w; c.gateb = gate_b;
  c.memory = memory; c.mask = mask;
  c.X = X; c.emoB = emoB; c.pm = pm;
  c.ahB = ahB; c.ctxB = ctxB; c.hA0 = hA0; c.hA1 = hA1; c.hD0 = hD0; c.hD1 = hD1;
  c.ahf = ahf; c.dhf = dhf; c.ctxf = ctxf; c.cA = cA; c.cD = cD;
  c.aw = aw; c.cum = cum; c.loc = loc;
  c.omel = omel; c.ogate = ogate; c.oalign = oalign;
  c.bar = bar;

  decoder_persistent<<<NBLK, 256, 0, stream>>>(c);
}
